// CausalSelfAttention_80410377715786
// MI455X (gfx1250) — matrix operand order repaired and checked by reading, not compiled, not run on hardware
//
#include <hip/hip_runtime.h>
#include <hip/hip_bf16.h>
#include <math.h>

typedef __attribute__((ext_vector_type(16))) _Float16 v16h;
typedef __attribute__((ext_vector_type(8)))  _Float16 v8h;
typedef __attribute__((ext_vector_type(8)))  float    v8f;
typedef _Float16 h16;

#define S_LEN   2048
#define DMODEL  512
#define NHEADS  8
#define HDIM    64
#define BATCH   4
#define NTOK    (BATCH * S_LEN)            // 8192

// ---------------------------------------------------------------------------
// WMMA 16x16x32 f16 fragment layouts (CDNA5 wave32); A/B use this code's own contiguous K order, not ISA table order:
//  A 16x32 : lane&15 = row M; element i -> K = hi*16 + i   (16 contiguous K)
//  B 32x16 : lane&15 = col N; element i -> K = hi*16 + i   (16 contiguous K)
//  C 16x16 : VGPR v -> row v + 8*hi; col = lane&15
// ---------------------------------------------------------------------------

__device__ inline v16h load_a_f16(const h16* rowp, int hi) {
  v8h lo = *(const v8h*)(rowp + hi * 16);
  v8h up = *(const v8h*)(rowp + hi * 16 + 8);
  v16h a;
#pragma unroll
  for (int i = 0; i < 8; ++i) { a[i] = lo[i]; a[8 + i] = up[i]; }
  return a;
}

__device__ inline void load_kb_frags(const h16* K, int j0, int lo, int hi,
                                     v16h kb[2][2]) {
#pragma unroll
  for (int t = 0; t < 2; ++t) {
    const h16* krow = K + (size_t)(j0 + t * 16 + lo) * HDIM;
#pragma unroll
    for (int kh = 0; kh < 2; ++kh)
      kb[t][kh] = *(const v16h*)(krow + kh * 32 + hi * 16);
  }
}

// ---------------------------------------------------------------------------
// Pre-pass: fp32 -> f16 conversions (one-time, L2-resident)
// ---------------------------------------------------------------------------
__global__ void __launch_bounds__(256)
cvt_src_kernel(const float* __restrict__ src, h16* __restrict__ dst) {
  const size_t idx = ((size_t)blockIdx.x * 256 + threadIdx.x) * 8;
  float4 f0 = *(const float4*)(src + idx);
  float4 f1 = *(const float4*)(src + idx + 4);
  v8h o;
  o[0] = (h16)f0.x; o[1] = (h16)f0.y; o[2] = (h16)f0.z; o[3] = (h16)f0.w;
  o[4] = (h16)f1.x; o[5] = (h16)f1.y; o[6] = (h16)f1.z; o[7] = (h16)f1.w;
  *(v8h*)(dst + idx) = o;
}

__global__ void __launch_bounds__(256)
cvt_w_kernel(const float* __restrict__ Wq, const float* __restrict__ Wk,
             const float* __restrict__ Wv, h16* __restrict__ wt) {
  const float* W = (blockIdx.y == 0) ? Wq : (blockIdx.y == 1) ? Wk : Wv;
  h16* dst = wt + (size_t)blockIdx.y * DMODEL * DMODEL;
  const int idx = blockIdx.x * 256 + threadIdx.x;   // 0 .. 512*512-1
  const int c = idx >> 9;
  const int r = idx & (DMODEL - 1);
  dst[(size_t)c * DMODEL + r] = (h16)W[(size_t)r * DMODEL + c];   // W^T, f16
}

// ---------------------------------------------------------------------------
// Phase 1: QKV projection GEMM, f16 in / f16 out, fp32 accumulate.
// Wave computes a 16x64 tile; explicit double-buffering of A/B fragments so
// loads for step k+1 overlap the 4 WMMAs of step k.
// ---------------------------------------------------------------------------
__global__ void __launch_bounds__(128)
qkv_gemm_kernel(const h16* __restrict__ srch, const h16* __restrict__ wt,
                const float* __restrict__ bq, const float* __restrict__ bk,
                const float* __restrict__ bv,
                h16* __restrict__ qws, h16* __restrict__ kws, h16* __restrict__ vws) {
  const int lane = threadIdx.x & 31;
  const int wave = threadIdx.x >> 5;
  const int hi   = lane >> 4;
  const int lo   = lane & 15;

  const int row0 = (blockIdx.x * 4 + wave) * 16;    // token tile (0..8191)
  const int colq = blockIdx.y;                      // 64-wide column quad = head
  const int z    = blockIdx.z;                      // 0=Q 1=K 2=V

  const h16*   Wt   = wt + (size_t)z * DMODEL * DMODEL;    // [n][k] f16
  const float* bias = (z == 0) ? bq : (z == 1) ? bk : bv;

  const h16* arow = srch + (size_t)(row0 + lo) * DMODEL;
  const h16* bcol[4];
#pragma unroll
  for (int t = 0; t < 4; ++t)
    bcol[t] = Wt + (size_t)(colq * 64 + t * 16 + lo) * DMODEL + hi * 16;

  // ---- prologue: load k-step 0 --------------------------------------------
  v16h a_cur = load_a_f16(arow, hi);
  v16h b_cur[4];
#pragma unroll
  for (int t = 0; t < 4; ++t) b_cur[t] = *(const v16h*)(bcol[t]);

  v8f acc[4] = {};
#pragma unroll 2
  for (int k0 = 0; k0 < DMODEL - 32; k0 += 32) {
    // prefetch k-step k0+32 while current WMMAs execute
    v16h a_nxt = load_a_f16(arow + k0 + 32, hi);
    v16h b_nxt[4];
#pragma unroll
    for (int t = 0; t < 4; ++t) b_nxt[t] = *(const v16h*)(bcol[t] + k0 + 32);

#pragma unroll
    for (int t = 0; t < 4; ++t)
      acc[t] = __builtin_amdgcn_wmma_f32_16x16x32_f16(false, a_cur, false, b_cur[t],
                                                      (short)0, acc[t], false, false);
    a_cur = a_nxt;
#pragma unroll
    for (int t = 0; t < 4; ++t) b_cur[t] = b_nxt[t];
  }
  // ---- epilogue: last k-step ---------------------------------------------
#pragma unroll
  for (int t = 0; t < 4; ++t)
    acc[t] = __builtin_amdgcn_wmma_f32_16x16x32_f16(false, a_cur, false, b_cur[t],
                                                    (short)0, acc[t], false, false);

#pragma unroll
  for (int t = 0; t < 4; ++t) {
    const int   colg = colq * 64 + t * 16 + lo;
    const float bsc  = bias[colg];
    const int   h    = colg >> 6;                 // head (== colq)
    const int   d    = colg & 63;
#pragma unroll
    for (int v = 0; v < 8; ++v) {
      const int tok = row0 + v + 8 * hi;
      const int b_  = tok >> 11;
      const int s_  = tok & (S_LEN - 1);
      const h16 val = (h16)(acc[t][v] + bsc);
      if (z == 2)      // V stored transposed: [B,H,hd,S]
        vws[(((size_t)b_ * NHEADS + h) * HDIM + d) * S_LEN + s_] = val;
      else if (z == 1)
        kws[(((size_t)b_ * NHEADS + h) * S_LEN + s_) * HDIM + d] = val;
      else
        qws[(((size_t)b_ * NHEADS + h) * S_LEN + s_) * HDIM + d] = val;
    }
  }
}

// ---------------------------------------------------------------------------
// Phase 2: flash attention (no mask). One wave = 16 query rows of one (b,h).
// Keys streamed in chunks of 32. V fragments load at chunk top (overlap the
// softmax VALU block); K fragments double-buffered across chunks.
// ---------------------------------------------------------------------------
__global__ void __launch_bounds__(128)
attn_kernel(const h16* __restrict__ qws, const h16* __restrict__ kws,
            const h16* __restrict__ vws, float* __restrict__ out) {
  __shared__ h16 pshare[4][16][32];   // per-wave P-transpose slab (4 KB)

  const int lane = threadIdx.x & 31;
  const int wave = threadIdx.x >> 5;
  const int hi   = lane >> 4;
  const int lo   = lane & 15;

  const int bh = blockIdx.x;          // 0..31
  const int b_ = bh >> 3;
  const int h_ = bh & 7;
  const int q0 = (blockIdx.y * 4 + wave) * 16;

  const h16* Q  = qws + (size_t)bh * S_LEN * HDIM;
  const h16* K  = kws + (size_t)bh * S_LEN * HDIM;
  const h16* Vt = vws + (size_t)bh * HDIM * S_LEN;   // [hd][S]

  const h16* qrow = Q + (size_t)(q0 + lo) * HDIM;
  v16h qa[2];
#pragma unroll
  for (int kh = 0; kh < 2; ++kh) qa[kh] = load_a_f16(qrow + kh * 32, hi);

  v8f   o[4] = {};
  float m[8], l[8];
#pragma unroll
  for (int v = 0; v < 8; ++v) { m[v] = -INFINITY; l[v] = 0.0f; }

  const float ascale = 0.125f * 1.4426950408889634f;   // 1/sqrt(64) * log2(e)

  v16h kb_cur[2][2];
  load_kb_frags(K, 0, lo, hi, kb_cur);                 // prologue chunk 0

  for (int j0 = 0; j0 < S_LEN; j0 += 32) {
    // ---- V fragments for THIS chunk: issue early, consumed after softmax --
    v16h vb[4];
#pragma unroll
    for (int t = 0; t < 4; ++t)
      vb[t] = *(const v16h*)(Vt + (size_t)(t * 16 + lo) * S_LEN + j0 + hi * 16);

    // ---- scores: 16 queries x 32 keys (two 16x16 C tiles) ----------------
    v8f sc[2];
#pragma unroll
    for (int t = 0; t < 2; ++t) {
      v8f a2 = {};
      a2 = __builtin_amdgcn_wmma_f32_16x16x32_f16(false, qa[0], false, kb_cur[t][0],
                                                  (short)0, a2, false, false);
      a2 = __builtin_amdgcn_wmma_f32_16x16x32_f16(false, qa[1], false, kb_cur[t][1],
                                                  (short)0, a2, false, false);
      sc[t] = a2;
    }

    // ---- prefetch NEXT chunk's K fragments (wrap keeps it in-bounds) -----
    v16h kb_nxt[2][2];
    load_kb_frags(K, (j0 + 32) & (S_LEN - 1), lo, hi, kb_nxt);

    // ---- online softmax (exp2 domain) ------------------------------------
#pragma unroll
    for (int v = 0; v < 8; ++v) {
      const float s0 = sc[0][v] * ascale;
      const float s1 = sc[1][v] * ascale;
      float mx = fmaxf(s0, s1);
#pragma unroll
      for (int msk = 1; msk < 16; msk <<= 1)
        mx = fmaxf(mx, __shfl_xor(mx, msk, 32));
      const float mnew = fmaxf(m[v], mx);
      const float p0 = __builtin_exp2f(s0 - mnew);
      const float p1 = __builtin_exp2f(s1 - mnew);
      float sum = p0 + p1;
#pragma unroll
      for (int msk = 1; msk < 16; msk <<= 1)
        sum += __shfl_xor(sum, msk, 32);
      const float corr = __builtin_exp2f(m[v] - mnew);
      l[v] = l[v] * corr + sum;
      m[v] = mnew;
#pragma unroll
      for (int t = 0; t < 4; ++t) o[t][v] *= corr;
      pshare[wave][v + 8 * hi][lo]      = (h16)p0;   // C-layout -> LDS
      pshare[wave][v + 8 * hi][16 + lo] = (h16)p1;
    }

    // ---- P transposed through LDS into A-fragment layout -----------------
    v8h plo = *(const v8h*)&pshare[wave][lo][hi * 16];
    v8h pup = *(const v8h*)&pshare[wave][lo][hi * 16 + 8];
    v16h pa;
#pragma unroll
    for (int i = 0; i < 8; ++i) { pa[i] = plo[i]; pa[8 + i] = pup[i]; }

    // ---- O += P(16x32) @ V(32x64) ----------------------------------------
#pragma unroll
    for (int t = 0; t < 4; ++t)
      o[t] = __builtin_amdgcn_wmma_f32_16x16x32_f16(false, pa, false, vb[t],
                                                    (short)0, o[t], false, false);

#pragma unroll
    for (int t = 0; t < 2; ++t) {
      kb_cur[t][0] = kb_nxt[t][0];
      kb_cur[t][1] = kb_nxt[t][1];
    }
  }

  // ---- finalize: divide by row sum, write fp32 [B, S, H, hd] -------------
#pragma unroll
  for (int v = 0; v < 8; ++v) {
    const float inv = 1.0f / l[v];
    const int   s_  = q0 + v + 8 * hi;
#pragma unroll
    for (int t = 0; t < 4; ++t) {
      out[(((size_t)b_ * S_LEN + s_) * NHEADS + h_) * HDIM + t * 16 + lo] =
          o[t][v] * inv;
    }
  }
}

// ---------------------------------------------------------------------------
extern "C" void kernel_launch(void* const* d_in, const int* in_sizes, int n_in,
                              void* d_out, int out_size, void* d_ws, size_t ws_size,
                              hipStream_t stream) {
  (void)in_sizes; (void)n_in; (void)out_size; (void)ws_size;

  const float* src = (const float*)d_in[0];
  const float* Wq  = (const float*)d_in[1];
  const float* bq  = (const float*)d_in[2];
  const float* Wk  = (const float*)d_in[3];
  const float* bk  = (const float*)d_in[4];
  const float* Wv  = (const float*)d_in[5];
  const float* bv  = (const float*)d_in[6];

  const size_t src_elems = (size_t)NTOK * DMODEL;            // 4 Mi
  const size_t w_elems   = (size_t)DMODEL * DMODEL;          // 256 Ki
  h16* srch = (h16*)d_ws;                                    // 8 MB
  h16* wt   = srch + src_elems;                              // 1.5 MB (x3)
  h16* qws  = wt + 3 * w_elems;                              // 8 MB
  h16* kws  = qws + src_elems;                               // 8 MB
  h16* vws  = kws + src_elems;                               // 8 MB

  // Pre-pass conversions
  cvt_src_kernel<<<dim3(src_elems / (256 * 8)), dim3(256), 0, stream>>>(src, srch);
  cvt_w_kernel<<<dim3(w_elems / 256, 3), dim3(256), 0, stream>>>(Wq, Wk, Wv, wt);

  // Phase 1: 512 row tiles / 4 waves per block; 8 column quads (heads); 3 mats
  qkv_gemm_kernel<<<dim3(128, 8, 3), dim3(128), 0, stream>>>(
      srch, wt, bq, bk, bv, qws, kws, vws);

  // Phase 2: 32 (b,h) x 32 blocks x 4 waves = 4096 query tiles of 16
  attn_kernel<<<dim3(32, 32), dim3(128), 0, stream>>>(qws, kws, vws, (float*)d_out);
}
